// GPT2_74826920230968
// MI455X (gfx1250) — matrix operand order repaired and checked by reading, not compiled, not run on hardware
//
#include <hip/hip_runtime.h>
#include <math.h>

// ---------------------------------------------------------------------------
// GPT-2 forward for MI455X (gfx1250)
//   - all matmuls on v_wmma_f32_16x16x32_bf16
//   - weights/activations pre-converted to bf16 (no conversions in hot loops)
//   - GEMM tile staging via Tensor Data Mover (tensor_load_to_lds) when
//     available, else GLOBAL_LOAD_ASYNC_TO_LDS_B128, else plain LDS stores
//   - softmax row reductions via DPP (no ds_bpermute)
// ---------------------------------------------------------------------------

#define BB   2
#define SS   1024
#define DD   768
#define HH   12
#define DHH  64
#define FFD  3072
#define LL   12
#define MM   (BB * SS)          // 2048 rows

typedef __bf16 v16bf __attribute__((ext_vector_type(16)));
typedef float  v8f   __attribute__((ext_vector_type(8)));
typedef int    v4i_  __attribute__((vector_size(16)));
typedef unsigned int v4u_ __attribute__((vector_size(16)));
typedef int    v8i_  __attribute__((vector_size(32)));

union Frag16 {
    v16bf        v;
    unsigned int w[8];
};

// RNE f32->bf16 (only in once-per-launch conversion / epilogues)
__device__ __forceinline__ unsigned short f2bf(float a) {
    unsigned int u = __float_as_uint(a);
    u += 0x7FFFu + ((u >> 16) & 1u);
    return (unsigned short)(u >> 16);
}
__device__ __forceinline__ unsigned int pack2bf(float a, float b) {
    unsigned int ua = __float_as_uint(a);
    ua += 0x7FFFu + ((ua >> 16) & 1u);
    unsigned int ub = __float_as_uint(b);
    ub += 0x7FFFu + ((ub >> 16) & 1u);
    return (ua >> 16) | (ub & 0xFFFF0000u);
}

// ---- async copy global(16B) -> LDS (fallback staging path) -----------------
#if __has_builtin(__builtin_amdgcn_global_load_async_to_lds_b128)
#define HAVE_ASYNC_LDS 1
#endif

__device__ __forceinline__ void cp16_g2l(const void* g, void* l) {
#ifdef HAVE_ASYNC_LDS
    __builtin_amdgcn_global_load_async_to_lds_b128(
        (__attribute__((address_space(1))) v4i_*)(unsigned long long)(uintptr_t)g,
        (__attribute__((address_space(3))) v4i_*)(unsigned int)(uintptr_t)l,
        0, 0);
#else
    *(uint4*)l = *(const uint4*)g;
#endif
}
__device__ __forceinline__ void wait_async_lds() {
#ifdef HAVE_ASYNC_LDS
#if __has_builtin(__builtin_amdgcn_s_wait_asynccnt)
    __builtin_amdgcn_s_wait_asynccnt(0);
#else
    asm volatile("s_wait_asynccnt 0" ::: "memory");
#endif
#endif
}

// ---- Tensor Data Mover: 2D tile DMA global -> LDS --------------------------
#if __has_builtin(__builtin_amdgcn_tensor_load_to_lds)
#define HAVE_TDM 1

// tile_w elems contiguous (dim0), tile_h rows (dim1); 2-byte elements.
// pad codes per D# group1: interval 0=2dw..5=64dw ; amount 0=1dw..127=128dw
__device__ __forceinline__ void tdm_load_2d(const void* gaddr, unsigned lds_off,
                                            unsigned tensor_w, unsigned tensor_h,
                                            unsigned long long stride_elems,
                                            unsigned tile_w, unsigned tile_h,
                                            int pad_interval, int pad_amount) {
    unsigned long long ga = (unsigned long long)(uintptr_t)gaddr;
    v4u_ g0;
    g0[0] = 1u;                                   // count=1, user mode
    g0[1] = lds_off;                              // lds_addr (bytes)
    g0[2] = (unsigned)(ga & 0xFFFFFFFFu);         // global_addr[31:0]
    g0[3] = (unsigned)((ga >> 32) & 0x1FFFFFFu)   // global_addr[56:32]
          | (2u << 30);                           // type = 2 ("image")
    v8i_ g1;
    unsigned w0 = (1u << 16);                     // data_size = 1 (2 bytes)
    if (pad_amount >= 0)
        w0 |= (1u << 20) | ((unsigned)pad_interval << 22)
            | ((unsigned)pad_amount << 25);
    g1[0] = (int)w0;
    g1[1] = (int)((tensor_w & 0xFFFFu) << 16);            // tensor_dim0[15:0]
    g1[2] = (int)(((tensor_w >> 16) & 0xFFFFu)            // tensor_dim0[31:16]
          | ((tensor_h & 0xFFFFu) << 16));                // tensor_dim1[15:0]
    g1[3] = (int)(((tensor_h >> 16) & 0xFFFFu)            // tensor_dim1[31:16]
          | ((tile_w & 0xFFFFu) << 16));                  // tile_dim0
    g1[4] = (int)(tile_h & 0xFFFFu);                      // tile_dim1
    g1[5] = (int)(unsigned)(stride_elems & 0xFFFFFFFFu);  // dim0_stride[31:0]
    g1[6] = (int)(unsigned)((stride_elems >> 32) & 0xFFFFu);
    g1[7] = 0;
    v4i_ z4 = {0, 0, 0, 0};
    v8i_ z8 = {0, 0, 0, 0, 0, 0, 0, 0};
    __builtin_amdgcn_tensor_load_to_lds(g0, g1, z4, z4, z8, 0);
}
__device__ __forceinline__ void wait_tensor0() {
#if __has_builtin(__builtin_amdgcn_s_wait_tensorcnt)
    __builtin_amdgcn_s_wait_tensorcnt(0);
#else
    asm volatile("s_wait_tensorcnt 0" ::: "memory");
#endif
}
#endif  // HAVE_TDM

// ---- DPP 16-lane row butterfly reductions ----------------------------------
#if __has_builtin(__builtin_amdgcn_update_dpp)
#define DPPSTEP(x, ctrl, OP)                                                   \
    {                                                                          \
        int _t = __builtin_amdgcn_update_dpp(0, __float_as_int(x), ctrl, 0xF,  \
                                             0xF, true);                       \
        x = OP(x, __int_as_float(_t));                                         \
    }
__device__ __forceinline__ float rowmax16(float x) {
    DPPSTEP(x, 0x0B1, fmaxf)   // quad_perm(1,0,3,2): xor1
    DPPSTEP(x, 0x04E, fmaxf)   // quad_perm(2,3,0,1): xor2
    DPPSTEP(x, 0x141, fmaxf)   // row_half_mirror
    DPPSTEP(x, 0x140, fmaxf)   // row_mirror
    return x;
}
__device__ __forceinline__ float fadd2(float a, float b) { return a + b; }
__device__ __forceinline__ float rowsum16(float x) {
    DPPSTEP(x, 0x0B1, fadd2)
    DPPSTEP(x, 0x04E, fadd2)
    DPPSTEP(x, 0x141, fadd2)
    DPPSTEP(x, 0x140, fadd2)
    return x;
}
#else
__device__ __forceinline__ float rowmax16(float x) {
    for (int off = 1; off < 16; off <<= 1) x = fmaxf(x, __shfl_xor(x, off, 32));
    return x;
}
__device__ __forceinline__ float rowsum16(float x) {
    for (int off = 1; off < 16; off <<= 1) x += __shfl_xor(x, off, 32);
    return x;
}
#endif

// ---------------------------------------------------------------------------
// x = a + b (elementwise, f32)
// ---------------------------------------------------------------------------
__global__ __launch_bounds__(256) void k_add(const float* __restrict__ a,
                                             const float* __restrict__ b,
                                             float* __restrict__ o, int n) {
    int i = blockIdx.x * 256 + threadIdx.x;
    if (i < n) o[i] = a[i] + b[i];
}

// ---------------------------------------------------------------------------
// f32 -> bf16 pair conversion (weights), optional scale folded in
// ---------------------------------------------------------------------------
__global__ __launch_bounds__(256) void k_cvt(const float* __restrict__ s,
                                             unsigned short* __restrict__ d,
                                             long long npairs, float sc) {
    long long i = (long long)blockIdx.x * 256 + threadIdx.x;
    if (i < npairs) {
        float a = s[2 * i] * sc;
        float b = s[2 * i + 1] * sc;
        ((unsigned int*)d)[i] = pack2bf(a, b);
    }
}

// ---------------------------------------------------------------------------
// LayerNorm rows of 768. OUTBF=1 -> bf16 output (feeds WMMA), else f32.
// ---------------------------------------------------------------------------
template <int OUTBF>
__global__ __launch_bounds__(256) void k_ln(const float* __restrict__ x,
                                            const float* __restrict__ g,
                                            const float* __restrict__ bt,
                                            void* __restrict__ outp) {
    __shared__ float red[256];
    const int row = blockIdx.x;
    const int tid = threadIdx.x;
    const float* xr = x + (size_t)row * DD;

    float v0 = xr[tid], v1 = xr[tid + 256], v2 = xr[tid + 512];
    red[tid] = v0 + v1 + v2;
    __syncthreads();
#pragma unroll
    for (int off = 128; off > 0; off >>= 1) {
        if (tid < off) red[tid] += red[tid + off];
        __syncthreads();
    }
    const float mean = red[0] * (1.0f / (float)DD);
    __syncthreads();

    float d0 = v0 - mean, d1 = v1 - mean, d2 = v2 - mean;
    red[tid] = d0 * d0 + d1 * d1 + d2 * d2;
    __syncthreads();
#pragma unroll
    for (int off = 128; off > 0; off >>= 1) {
        if (tid < off) red[tid] += red[tid + off];
        __syncthreads();
    }
    const float rs = rsqrtf(red[0] * (1.0f / (float)DD) + 0.02f);

    float o0 = d0 * rs * g[tid]       + bt[tid];
    float o1 = d1 * rs * g[tid + 256] + bt[tid + 256];
    float o2 = d2 * rs * g[tid + 512] + bt[tid + 512];
    if (OUTBF) {
        unsigned short* orow = (unsigned short*)outp + (size_t)row * DD;
        orow[tid] = f2bf(o0); orow[tid + 256] = f2bf(o1); orow[tid + 512] = f2bf(o2);
    } else {
        float* orow = (float*)outp + (size_t)row * DD;
        orow[tid] = o0; orow[tid + 256] = o1; orow[tid + 512] = o2;
    }
}

// ---------------------------------------------------------------------------
// GEMM: C[M,N] = act(A[M,K](bf16) * W[K,N](bf16) + bias*bscale (+ residual))
// Block 256 = 8 waves; tile 128x128, K-step 64 (16 WMMAs per stage).
// ACT: 0 none, 1 exact GELU, 2 residual add.  OUTBF: 1 -> bf16 C, 0 -> f32 C.
// ---------------------------------------------------------------------------
#define BK     64
#define APITCH 72    // 64 elems + 4-dword pad: 144B rows, conflict-free
#define BPITCH 136   // 128 elems + 4-dword pad: 272B rows

template <int ACT, int OUTBF>
__global__ __launch_bounds__(256) void k_gemm(const unsigned short* __restrict__ A,
                                              const unsigned short* __restrict__ W,
                                              const float* __restrict__ bias,
                                              float bscale,
                                              const float* __restrict__ res,
                                              void* __restrict__ Cp,
                                              int M, int N, int K) {
    __shared__ unsigned short As[128 * APITCH];  // [m][k]
    __shared__ unsigned short Bs[BK * BPITCH];   // [k][n] row-major

    const int tid  = threadIdx.x;
    const int lane = tid & 31;
    const int wid  = tid >> 5;
    const int wr   = wid >> 1;           // 0..3 -> 32 rows each
    const int wc   = wid & 1;            // 0..1 -> 64 cols each
    const int half = lane >> 4;
    const int l15  = lane & 15;
    const int m0   = blockIdx.y * 128;
    const int n0   = blockIdx.x * 128;

    v8f acc[2][4] = {};

#ifndef HAVE_TDM
    const int arow  = tid >> 1;          // 0..127
    const int akoff = (tid & 1) * 32;    // 0 / 32 elems
    const int bkrow = tid >> 2;          // 0..63
    const int bnoff = (tid & 3) * 32;    // 0..96 elems
    unsigned short* asl = As + arow * APITCH + akoff;
    unsigned short* bsl = Bs + bkrow * BPITCH + bnoff;
#endif

    for (int k0 = 0; k0 < K; k0 += BK) {
#ifdef HAVE_TDM
        if (wid == 0) {
            // A tile: 128 rows x 64 elems; pad 4dw every 32dw -> pitch 72
            tdm_load_2d(A + (size_t)m0 * K + k0,
                        (unsigned)(uintptr_t)(void*)As,
                        (unsigned)K, (unsigned)M, (unsigned long long)K,
                        BK, 128, /*interval 32dw*/ 4, /*amount 4dw*/ 3);
            // W tile: 64 rows x 128 elems; pad 4dw every 64dw -> pitch 136
            tdm_load_2d(W + (size_t)k0 * N + n0,
                        (unsigned)(uintptr_t)(void*)Bs,
                        (unsigned)N, (unsigned)K, (unsigned long long)N,
                        128, BK, /*interval 64dw*/ 5, /*amount 4dw*/ 3);
            wait_tensor0();
        }
#else
        {
            const unsigned short* ag = A + (size_t)(m0 + arow) * K + k0 + akoff;
            const unsigned short* wg = W + (size_t)(k0 + bkrow) * N + n0 + bnoff;
            cp16_g2l(ag,      asl);
            cp16_g2l(ag + 8,  asl + 8);
            cp16_g2l(ag + 16, asl + 16);
            cp16_g2l(ag + 24, asl + 24);
            cp16_g2l(wg,      bsl);
            cp16_g2l(wg + 8,  bsl + 8);
            cp16_g2l(wg + 16, bsl + 16);
            cp16_g2l(wg + 24, bsl + 24);
            wait_async_lds();
        }
#endif
        __syncthreads();

#pragma unroll
        for (int ks = 0; ks < 2; ++ks) {
            // A fragments (contiguous k = 16*half + s order, same as B) -- aligned b32 reads
            Frag16 af[2];
#pragma unroll
            for (int mi = 0; mi < 2; ++mi) {
                const unsigned short* base =
                    As + (wr * 32 + mi * 16 + l15) * APITCH + ks * 32;
                const int kb = half * 16;
#pragma unroll
                for (int vv = 0; vv < 4; ++vv) {
                    af[mi].w[vv]     = *(const unsigned int*)(base + kb + 2 * vv);
                    af[mi].w[vv + 4] = *(const unsigned int*)(base + kb + 8 + 2 * vv);
                }
            }
            // B fragments: transpose-on-read pair gather from [k][n] tile
#pragma unroll
            for (int ni = 0; ni < 4; ++ni) {
                Frag16 bf;
                const int n = wc * 64 + ni * 16 + l15;
                const unsigned short* bb =
                    Bs + (ks * 32 + half * 16) * BPITCH + n;
#pragma unroll
                for (int vv = 0; vv < 8; ++vv) {
                    unsigned int lo = bb[(2 * vv) * BPITCH];
                    unsigned int hi = bb[(2 * vv + 1) * BPITCH];
                    bf.w[vv] = lo | (hi << 16);
                }
#pragma unroll
                for (int mi = 0; mi < 2; ++mi)
                    acc[mi][ni] = __builtin_amdgcn_wmma_f32_16x16x32_bf16(
                        false, af[mi].v, false, bf.v, (short)0, acc[mi][ni],
                        false, false);
            }
        }
        __syncthreads();
    }

    // epilogue
#pragma unroll
    for (int mi = 0; mi < 2; ++mi)
#pragma unroll
        for (int ni = 0; ni < 4; ++ni) {
            const int col = n0 + wc * 64 + ni * 16 + l15;
            const float bcol = bias[col] * bscale;
#pragma unroll
            for (int r = 0; r < 8; ++r) {
                const int row = m0 + wr * 32 + mi * 16 + r + 8 * half;
                float val = acc[mi][ni][r] + bcol;
                if (ACT == 1)
                    val = 0.5f * val * (1.0f + erff(val * 0.70710678118654752f));
                if (ACT == 2)
                    val += res[(size_t)row * N + col];
                if (OUTBF)
                    ((unsigned short*)Cp)[(size_t)row * N + col] = f2bf(val);
                else
                    ((float*)Cp)[(size_t)row * N + col] = val;
            }
        }
}

// ---------------------------------------------------------------------------
// Flash attention, one wave per (b, h, 16-query tile). q/k/v/o are bf16.
// Q pre-scaled by 1/sqrt(64) (folded into wq/bq). Causal via select-mask.
// ---------------------------------------------------------------------------
__global__ __launch_bounds__(32) void k_attn(const unsigned short* __restrict__ q,
                                             const unsigned short* __restrict__ k,
                                             const unsigned short* __restrict__ v,
                                             unsigned short* __restrict__ o) {
    __shared__ unsigned short Pld[16][32];

    const int lane = threadIdx.x;
    const int half = lane >> 4;
    const int l15  = lane & 15;
    const int b  = blockIdx.z;
    const int h  = blockIdx.y;
    const int qbase = blockIdx.x * 16;
    const size_t hoff = (size_t)h * DHH;

    // Q A-fragments: bf16 pairs are contiguous along dh -> aligned b32 loads
    Frag16 aQ[2];
    {
        const unsigned short* qp =
            q + ((size_t)(b * SS) + qbase + l15) * DD + hoff;
#pragma unroll
        for (int d = 0; d < 2; ++d)
#pragma unroll
            for (int vv = 0; vv < 8; ++vv) {
                const int kk = d * 32 + half * 16 + 2 * vv;
                aQ[d].w[vv] = *(const unsigned int*)(qp + kk);
            }
    }

    float mrun[8], lrun[8];
    v8f accO[4] = {};
#pragma unroll
    for (int r = 0; r < 8; ++r) { mrun[r] = -INFINITY; lrun[r] = 0.0f; }

    for (int kb = 0; kb < qbase + 16; kb += 32) {
        // K B-fragments: lane = key, K-dim = dh, pairs contiguous -> b32 loads
        Frag16 bK[2][2];
#pragma unroll
        for (int g = 0; g < 2; ++g) {
            const unsigned short* kp =
                k + ((size_t)(b * SS) + kb + g * 16 + l15) * DD + hoff;
#pragma unroll
            for (int d = 0; d < 2; ++d)
#pragma unroll
                for (int vv = 0; vv < 8; ++vv)
                    bK[g][d].w[vv] =
                        *(const unsigned int*)(kp + d * 32 + half * 16 + 2 * vv);
        }

        v8f sf[2];
#pragma unroll
        for (int g = 0; g < 2; ++g) {
            v8f z = {};
            z = __builtin_amdgcn_wmma_f32_16x16x32_bf16(false, aQ[0].v, false,
                                                        bK[g][0].v, (short)0, z,
                                                        false, false);
            sf[g] = __builtin_amdgcn_wmma_f32_16x16x32_bf16(false, aQ[1].v, false,
                                                            bK[g][1].v, (short)0, z,
                                                            false, false);
        }

        // causal mask + online softmax (row lives across a 16-lane half)
#pragma unroll
        for (int r = 0; r < 8; ++r) {
            const int row = qbase + r + 8 * half;
#pragma unroll
            for (int g = 0; g < 2; ++g) {
                const int col = kb + g * 16 + l15;
                sf[g][r] = (col > row) ? -1.0e30f : sf[g][r];
            }
            const float mx   = rowmax16(fmaxf(sf[0][r], sf[1][r]));
            const float mnew = fmaxf(mrun[r], mx);
            const float corr = expf(mrun[r] - mnew);
            const float p0 = expf(sf[0][r] - mnew);
            const float p1 = expf(sf[1][r] - mnew);
            sf[0][r] = p0;
            sf[1][r] = p1;
            const float s = rowsum16(p0 + p1);
            lrun[r] = lrun[r] * corr + s;
            mrun[r] = mnew;
#pragma unroll
            for (int d = 0; d < 4; ++d) accO[d][r] *= corr;
        }

        // P: store bf16 to LDS, read back as A-fragment pairs
#pragma unroll
        for (int r = 0; r < 8; ++r)
#pragma unroll
            for (int g = 0; g < 2; ++g)
                Pld[r + 8 * half][g * 16 + l15] =
                    (unsigned short)(__float_as_uint(sf[g][r]) >> 16);
        __syncthreads();
        Frag16 aP;
#pragma unroll
        for (int vv = 0; vv < 8; ++vv) {
            const int kk = half * 16 + 2 * vv;
            aP.w[vv] = *(const unsigned int*)(&Pld[l15][kk]);
        }
        __syncthreads();

        // P @ V: V B-fragment = column pair gather (2 x u16 + merge per reg)
#pragma unroll
        for (int d = 0; d < 4; ++d) {
            Frag16 bV;
            const unsigned short* vp =
                v + ((size_t)(b * SS) + kb) * DD + hoff + d * 16 + l15;
#pragma unroll
            for (int vv = 0; vv < 8; ++vv) {
                const int key = half * 16 + 2 * vv;
                unsigned int lo = vp[(size_t)key * DD];
                unsigned int hi = vp[(size_t)(key + 1) * DD];
                bV.w[vv] = lo | (hi << 16);
            }
            accO[d] = __builtin_amdgcn_wmma_f32_16x16x32_bf16(
                false, aP.v, false, bV.v, (short)0, accO[d], false, false);
        }
    }

    // normalize + bf16 store, (B,S,H,DH) flat
#pragma unroll
    for (int r = 0; r < 8; ++r) {
        const float inv = 1.0f / lrun[r];
        const int row = qbase + r + 8 * half;
        unsigned short* op = o + ((size_t)(b * SS) + row) * DD + hoff;
#pragma unroll
        for (int d = 0; d < 4; ++d)
            op[d * 16 + l15] = f2bf(accO[d][r] * inv);
    }
}

// ---------------------------------------------------------------------------
// Host orchestration
// ---------------------------------------------------------------------------
extern "C" void kernel_launch(void* const* d_in, const int* in_sizes, int n_in,
                              void* d_out, int out_size, void* d_ws, size_t ws_size,
                              hipStream_t stream) {
    (void)in_sizes; (void)n_in; (void)out_size; (void)ws_size;

    const float* inputs = (const float*)d_in[0];
    const float* pos    = (const float*)d_in[1];
    // d_in[2] = attention_mask (causal, applied analytically)
    const float* wq = (const float*)d_in[3];
    const float* bq = (const float*)d_in[4];
    const float* wk = (const float*)d_in[5];
    const float* bk = (const float*)d_in[6];
    const float* wv = (const float*)d_in[7];
    const float* bv = (const float*)d_in[8];
    const float* wo = (const float*)d_in[9];
    const float* bo = (const float*)d_in[10];
    const float* ln1g = (const float*)d_in[11];
    const float* ln1b = (const float*)d_in[12];
    const float* ln2g = (const float*)d_in[13];
    const float* ln2b = (const float*)d_in[14];
    const float* w1 = (const float*)d_in[15];
    const float* b1 = (const float*)d_in[16];
    const float* w2 = (const float*)d_in[17];
    const float* b2 = (const float*)d_in[18];
    const float* lnfg = (const float*)d_in[19];
    const float* lnfb = (const float*)d_in[20];

    const size_t MD  = (size_t)MM * DD;
    const size_t WQN = (size_t)LL * DD * DD;    // elems per D x D tensor group
    const size_t W1N = (size_t)LL * DD * FFD;   // elems per D x FF tensor group

    unsigned short* p = (unsigned short*)d_ws;
    unsigned short* wqb = p; p += WQN;
    unsigned short* wkb = p; p += WQN;
    unsigned short* wvb = p; p += WQN;
    unsigned short* wob = p; p += WQN;
    unsigned short* w1b = p; p += W1N;
    unsigned short* w2b = p; p += W1N;
    float* x = (float*)p;                       // MM x DD f32 residual stream
    unsigned short* hbuf = (unsigned short*)(x + MD);
    unsigned short* qb   = hbuf + MD;
    unsigned short* kbuf = qb + MD;
    unsigned short* vbuf = kbuf + MD;
    unsigned short* atn  = vbuf + MD;
    unsigned short* ffb  = atn + MD;            // MM x FFD

    const dim3 blk256(256), blk32(32);
    const dim3 gLN(MM);
    const dim3 g768(DD / 128, MM / 128);
    const dim3 g3072(FFD / 128, MM / 128);
    const dim3 gAttn(SS / 16, HH, BB);

    // one-shot weight conversion f32 -> bf16 (1/sqrt(DH) folded into wq)
    auto cvt = [&](const float* s, unsigned short* d, size_t n, float sc) {
        long long np = (long long)(n / 2);
        k_cvt<<<dim3((unsigned)((np + 255) / 256)), blk256, 0, stream>>>(s, d, np, sc);
    };
    cvt(wq, wqb, WQN, 0.125f);
    cvt(wk, wkb, WQN, 1.0f);
    cvt(wv, wvb, WQN, 1.0f);
    cvt(wo, wob, WQN, 1.0f);
    cvt(w1, w1b, W1N, 1.0f);
    cvt(w2, w2b, W1N, 1.0f);

    k_add<<<dim3((unsigned)((MD + 255) / 256)), blk256, 0, stream>>>(
        inputs, pos, x, (int)MD);

    for (int i = 0; i < LL; ++i) {
        const size_t wOff  = (size_t)i * DD * DD;
        const size_t bOff  = (size_t)i * DD;
        const size_t w1Off = (size_t)i * DD * FFD;
        const size_t b1Off = (size_t)i * FFD;

        k_ln<1><<<gLN, blk256, 0, stream>>>(x, ln1g + bOff, ln1b + bOff, hbuf);

        k_gemm<0, 1><<<g768, blk256, 0, stream>>>(hbuf, wqb + wOff, bq + bOff,
                                                  0.125f, nullptr, qb, MM, DD, DD);
        k_gemm<0, 1><<<g768, blk256, 0, stream>>>(hbuf, wkb + wOff, bk + bOff,
                                                  1.0f, nullptr, kbuf, MM, DD, DD);
        k_gemm<0, 1><<<g768, blk256, 0, stream>>>(hbuf, wvb + wOff, bv + bOff,
                                                  1.0f, nullptr, vbuf, MM, DD, DD);

        k_attn<<<gAttn, blk32, 0, stream>>>(qb, kbuf, vbuf, atn);

        // x = attn @ wo + bo + x
        k_gemm<2, 0><<<g768, blk256, 0, stream>>>(atn, wob + wOff, bo + bOff,
                                                  1.0f, x, x, MM, DD, DD);

        k_ln<1><<<gLN, blk256, 0, stream>>>(x, ln2g + bOff, ln2b + bOff, hbuf);

        // ff = gelu(h @ w1 + b1)   (bf16 out, feeds next GEMM)
        k_gemm<1, 1><<<g3072, blk256, 0, stream>>>(hbuf, w1b + w1Off, b1 + b1Off,
                                                   1.0f, nullptr, ffb, MM, FFD, DD);
        // x = ff @ w2 + b2 + x
        k_gemm<2, 0><<<g768, blk256, 0, stream>>>(ffb, w2b + w1Off, b2 + bOff,
                                                  1.0f, x, x, MM, DD, FFD);
    }

    k_ln<0><<<gLN, blk256, 0, stream>>>(x, lnfg, lnfb, (float*)d_out);
}
